// EGNNDecoder_23416161698074
// MI455X (gfx1250) — matrix operand order repaired and checked by reading, not compiled, not run on hardware
//
#include <hip/hip_runtime.h>

typedef __attribute__((ext_vector_type(16))) _Float16 v16h;
typedef __attribute__((ext_vector_type(8)))  float    v8f;

#define NN    1024
#define NDEFN 512
#define EE    16384
#define TTOT  12
#define FFEAT 16
#define DENCD 32
#define HH    64
#define LL    3
#define EIN   133
#define EINP  160     // edge-mlp K padded to 5 x 32
#define KTRAJ 2

__device__ __forceinline__ float silu_f(float x) {
    return x / (1.0f + __expf(-x));
}

__device__ __forceinline__ v8f wmma_f16(v16h a, v16h b, v8f c) {
    // D = A(16x32,f16) * B(32x16,f16) + C(16x16,f32)
    return __builtin_amdgcn_wmma_f32_16x16x32_f16(false, a, false, b, (short)0, c, false, false);
}

// A fragment (16x32 MxK tile) from row-major f16 [16+][ld] source.
// Lane layout (ISA 7.12.2): lane m=lane&15, half=lane>>4; halfs [kB+half*8 .. +7]
// fill a[0..7], halfs [kB+16+half*8 .. +7] fill a[8..15].
__device__ __forceinline__ v16h load_A(const _Float16* S, int ld, int kBase, int lane) {
    int m = lane & 15, hf = lane >> 4;
    const _Float16* p = S + m * ld + kBase + hf * 8;
    v16h a;
#pragma unroll
    for (int j = 0; j < 8; ++j) { a[j] = p[j]; a[8 + j] = p[16 + j]; }
    return a;
}

__device__ __forceinline__ v16h load_A_f32(const float* S, int ld, int kBase, int lane) {
    int m = lane & 15, hf = lane >> 4;
    const float* p = S + m * ld + kBase + hf * 8;
    v16h a;
#pragma unroll
    for (int j = 0; j < 8; ++j) { a[j] = (_Float16)p[j]; a[8 + j] = (_Float16)p[16 + j]; }
    return a;
}

// B fragment (32x16 KxN tile) from TRANSPOSED f16 weight WT[n][k] (row length ldK).
// half=lane>>4: K=kB+half*8..+7 fill b[0..7], K=kB+16+half*8..+7 fill b[8..15].
__device__ __forceinline__ v16h load_B(const _Float16* WT, int ldK, int nBase, int kBase, int lane) {
    int n = nBase + (lane & 15), hf = lane >> 4;
    const _Float16* p = WT + n * ldK + kBase + hf * 8;
    v16h b;
#pragma unroll
    for (int j = 0; j < 8; ++j) { b[j] = p[j]; b[8 + j] = p[16 + j]; }
    return b;
}

// ---------------- weight convert + transpose to f16 ----------------
__global__ void convert_weights_kernel(const float* ew1, const float* ew2, const float* cw1,
                                       const float* vw1, const float* nw1, const float* nw2,
                                       _Float16* e1T, _Float16* e2T, _Float16* c1T,
                                       _Float16* v1T, _Float16* n1T, _Float16* n2T) {
    int i = blockIdx.x * blockDim.x + threadIdx.x;
    const int S0 = LL * HH * EINP;   // 30720
    const int S1 = LL * HH * HH;     // 12288
    const int S4 = LL * HH * 128;    // 24576
    if (i < S0) {
        int l = i / (HH * EINP), r = i % (HH * EINP);
        int n = r / EINP, k = r % EINP;
        e1T[i] = (_Float16)((k < EIN) ? ew1[(l * EIN + k) * HH + n] : 0.0f);
        return;
    }
    i -= S0;
    if (i < S1) { int l = i/(HH*HH), r = i%(HH*HH), n = r/HH, k = r%HH;
        e2T[i] = (_Float16)ew2[(l * HH + k) * HH + n]; return; }
    i -= S1;
    if (i < S1) { int l = i/(HH*HH), r = i%(HH*HH), n = r/HH, k = r%HH;
        c1T[i] = (_Float16)cw1[(l * HH + k) * HH + n]; return; }
    i -= S1;
    if (i < S1) { int l = i/(HH*HH), r = i%(HH*HH), n = r/HH, k = r%HH;
        v1T[i] = (_Float16)vw1[(l * HH + k) * HH + n]; return; }
    i -= S1;
    if (i < S4) { int l = i/(HH*128), r = i%(HH*128), n = r/128, k = r%128;
        n1T[i] = (_Float16)nw1[(l * 128 + k) * HH + n]; return; }
    i -= S4;
    if (i < S1) { int l = i/(HH*HH), r = i%(HH*HH), n = r/HH, k = r%HH;
        n2T[i] = (_Float16)nw2[(l * HH + k) * HH + n]; return; }
}

__global__ void zero_kernel(float* p, int n) {
    int i = blockIdx.x * blockDim.x + threadIdx.x;
    if (i < n) p[i] = 0.0f;
}

__global__ void deg_kernel(const int* eidx, float* deg) {
    int e = blockIdx.x * blockDim.x + threadIdx.x;
    if (e < EE) atomicAdd(&deg[eidx[e]], 1.0f);
}

// hh0 = concat(h[traj], enc) @ emb_in_w + b   (once per trajectory; reused every step)
__global__ void hh0_kernel(const float* h, const float* enc, const float* embw, const float* embb,
                           const int* tt, float* hh0, _Float16* hh0h) {
    int i = blockIdx.x * blockDim.x + threadIdx.x;
    if (i >= KTRAJ * NN * HH) return;
    int k = i / (NN * HH), r = i % (NN * HH);
    int n = r / HH, j = r % HH;
    int traj = tt[k];
    float s = embb[j];
    const float* hr = h + (traj * NN + n) * FFEAT;
    for (int f = 0; f < FFEAT; ++f) s += hr[f] * embw[f * HH + j];
    for (int d = 0; d < DENCD; ++d) s += enc[d] * embw[(FFEAT + d) * HH + j];
    hh0[i] = s;
    hh0h[i] = (_Float16)s;
}

__global__ void traj_init_kernel(const float* x, const float* v, const int* tt, int k,
                                 float* xc, float* vc, float* out) {
    int i = blockIdx.x * blockDim.x + threadIdx.x;
    if (i >= NN * 3) return;
    int traj = tt[k];
    int base = traj * TTOT * NN * 3;   // x[traj][0]
    float xv = x[base + i];
    xc[i] = xv;
    vc[i] = v[base + i];
    if (i < NDEFN * 3) out[(k * TTOT) * NDEFN * 3 + i] = xv;
}

__global__ void step_start_kernel(const float* hh0, const _Float16* hh0h, int k,
                                  float* hh, _Float16* hhh, const float* xc, float* xstart) {
    int i = blockIdx.x * blockDim.x + threadIdx.x;
    if (i < NN * HH) { hh[i] = hh0[k * NN * HH + i]; hhh[i] = hh0h[k * NN * HH + i]; }
    if (i < NN * 3)  xstart[i] = xc[i];
}

__global__ void step_end_kernel(const float* xc, const float* xstart, int k, int t,
                                float* vc, float* out) {
    int i = blockIdx.x * blockDim.x + threadIdx.x;
    if (i >= NN * 3) return;
    vc[i] = xc[i] - xstart[i];
    if (i < NDEFN * 3) out[(k * TTOT + t) * NDEFN * 3 + i] = xc[i];
}

// ------------- edge kernel: one wave = 16 edges; 4 waves/block -------------
__global__ __launch_bounds__(128) void edge_kernel(
    const float* x, const _Float16* hh16, const float* ef, const int* eidx,
    const _Float16* e1T, const float* eb1, const _Float16* e2T, const float* eb2,
    const _Float16* c1T, const float* cb1, const float* cw2,
    float* agg, float* hagg) {
    __shared__ _Float16 Sin[4][16][EINP];   // edge-mlp input rows (f16, zero padded)
    __shared__ _Float16 Sm[4][16][HH];      // A-fragment round-trip buffer
    __shared__ float    Sphi[4][16][16];
    __shared__ float    Sdiff[4][16][3];
    __shared__ int      Srow[4][16];

    int wave = threadIdx.x >> 5;
    int lane = threadIdx.x & 31;
    int eBase = (blockIdx.x * 4 + wave) * 16;
    int r = lane & 15, part = lane >> 4;
    int e = eBase + r;
    int ri = eidx[e];
    int ci = eidx[EE + e];

    // stage concat(hh[row], hh[col], radial, edge_feat, 0-pad) as f16
    const _Float16* hi = hh16 + ri * HH + part * 32;
    const _Float16* hj = hh16 + ci * HH + part * 32;
#pragma unroll
    for (int j = 0; j < 32; ++j) {
        Sin[wave][r][part * 32 + j]      = hi[j];
        Sin[wave][r][64 + part * 32 + j] = hj[j];
    }
    if (part == 0) {
        Srow[wave][r] = ri;
        float dx = x[ri * 3 + 0] - x[ci * 3 + 0];
        float dy = x[ri * 3 + 1] - x[ci * 3 + 1];
        float dz = x[ri * 3 + 2] - x[ci * 3 + 2];
        Sdiff[wave][r][0] = dx; Sdiff[wave][r][1] = dy; Sdiff[wave][r][2] = dz;
        Sin[wave][r][128] = (_Float16)(dx * dx + dy * dy + dz * dz);
        Sin[wave][r][129] = (_Float16)ef[e * 4 + 0];
        Sin[wave][r][130] = (_Float16)ef[e * 4 + 1];
        Sin[wave][r][131] = (_Float16)ef[e * 4 + 2];
        Sin[wave][r][132] = (_Float16)ef[e * 4 + 3];
#pragma unroll
        for (int kk = EIN; kk < EINP; ++kk) Sin[wave][r][kk] = (_Float16)0.0f;
    }
    __syncthreads();

    int nl = lane & 15, hf = lane >> 4;

    // GEMM1: [16x160] @ edge_w1 -> silu -> Sm
    v16h a1[5];
#pragma unroll
    for (int c = 0; c < 5; ++c) a1[c] = load_A(&Sin[wave][0][0], EINP, c * 32, lane);
#pragma unroll
    for (int nt = 0; nt < 4; ++nt) {
        v8f acc = {0.f, 0.f, 0.f, 0.f, 0.f, 0.f, 0.f, 0.f};
#pragma unroll
        for (int c = 0; c < 5; ++c)
            acc = wmma_f16(a1[c], load_B(e1T, EINP, nt * 16, c * 32, lane), acc);
        int n = nt * 16 + nl;
        float bb = eb1[n];
#pragma unroll
        for (int i = 0; i < 8; ++i)
            Sm[wave][i + 8 * hf][n] = (_Float16)silu_f(acc[i] + bb);
    }
    __syncthreads();

    // GEMM2: m = silu([16x64] @ edge_w2); hagg atomics; Sm <- m
    v16h a2[2];
    a2[0] = load_A(&Sm[wave][0][0], HH, 0, lane);
    a2[1] = load_A(&Sm[wave][0][0], HH, 32, lane);
#pragma unroll
    for (int nt = 0; nt < 4; ++nt) {
        v8f acc = {0.f, 0.f, 0.f, 0.f, 0.f, 0.f, 0.f, 0.f};
        acc = wmma_f16(a2[0], load_B(e2T, HH, nt * 16, 0, lane), acc);
        acc = wmma_f16(a2[1], load_B(e2T, HH, nt * 16, 32, lane), acc);
        int n = nt * 16 + nl;
        float bb = eb2[n];
#pragma unroll
        for (int i = 0; i < 8; ++i) {
            int m = i + 8 * hf;
            float vv = silu_f(acc[i] + bb);
            atomicAdd(&hagg[Srow[wave][m] * HH + n], vv);
            Sm[wave][m][n] = (_Float16)vv;
        }
    }
    __syncthreads();

    // GEMM3: phi = silu(m @ coord_w1 + b) @ coord_w2 ; agg += diff*phi (mean later)
    v16h a3[2];
    a3[0] = load_A(&Sm[wave][0][0], HH, 0, lane);
    a3[1] = load_A(&Sm[wave][0][0], HH, 32, lane);
    float pp[8] = {0.f, 0.f, 0.f, 0.f, 0.f, 0.f, 0.f, 0.f};
#pragma unroll
    for (int nt = 0; nt < 4; ++nt) {
        v8f acc = {0.f, 0.f, 0.f, 0.f, 0.f, 0.f, 0.f, 0.f};
        acc = wmma_f16(a3[0], load_B(c1T, HH, nt * 16, 0, lane), acc);
        acc = wmma_f16(a3[1], load_B(c1T, HH, nt * 16, 32, lane), acc);
        int n = nt * 16 + nl;
        float bb = cb1[n], w2 = cw2[n];
#pragma unroll
        for (int i = 0; i < 8; ++i) pp[i] += silu_f(acc[i] + bb) * w2;
    }
#pragma unroll
    for (int i = 0; i < 8; ++i) Sphi[wave][i + 8 * hf][nl] = pp[i];
    __syncthreads();
    if (part == 0) {
        float phi = 0.f;
#pragma unroll
        for (int j = 0; j < 16; ++j) phi += Sphi[wave][r][j];
        int node = Srow[wave][r];
        atomicAdd(&agg[node * 3 + 0], Sdiff[wave][r][0] * phi);
        atomicAdd(&agg[node * 3 + 1], Sdiff[wave][r][1] * phi);
        atomicAdd(&agg[node * 3 + 2], Sdiff[wave][r][2] * phi);
    }
}

// ------------- node kernel: one wave = 16 nodes; 4 waves/block -------------
__global__ __launch_bounds__(128) void node_kernel(
    _Float16* hh16, float* hh, const float* hagg, const float* agg, const float* deg,
    const float* vc, float* xc,
    const _Float16* v1T, const float* vb1, const float* vw2, const float* vb2,
    const _Float16* n1T, const float* nb1, const _Float16* n2T, const float* nb2) {
    __shared__ _Float16 Sm[4][16][HH];
    __shared__ float    Sphi[4][16][16];

    int wave = threadIdx.x >> 5;
    int lane = threadIdx.x & 31;
    int nodeBase = (blockIdx.x * 4 + wave) * 16;
    int nl = lane & 15, hf = lane >> 4;

    v16h ah[2];
    ah[0] = load_A(hh16 + nodeBase * HH, HH, 0, lane);
    ah[1] = load_A(hh16 + nodeBase * HH, HH, 32, lane);

    // velocity weight: velw = silu(hh @ vel_w1 + b1) @ vel_w2 + b2
    float pp[8] = {0.f, 0.f, 0.f, 0.f, 0.f, 0.f, 0.f, 0.f};
#pragma unroll
    for (int nt = 0; nt < 4; ++nt) {
        v8f acc = {0.f, 0.f, 0.f, 0.f, 0.f, 0.f, 0.f, 0.f};
        acc = wmma_f16(ah[0], load_B(v1T, HH, nt * 16, 0, lane), acc);
        acc = wmma_f16(ah[1], load_B(v1T, HH, nt * 16, 32, lane), acc);
        int n = nt * 16 + nl;
        float bb = vb1[n], w2 = vw2[n];
#pragma unroll
        for (int i = 0; i < 8; ++i) pp[i] += silu_f(acc[i] + bb) * w2;
    }
#pragma unroll
    for (int i = 0; i < 8; ++i) Sphi[wave][i + 8 * hf][nl] = pp[i];
    __syncthreads();
    if (hf == 0) {
        float velw = vb2[0];
#pragma unroll
        for (int j = 0; j < 16; ++j) velw += Sphi[wave][nl][j];
        int node = nodeBase + nl;
        float dg = deg[node]; if (dg < 1.0f) dg = 1.0f;
#pragma unroll
        for (int d = 0; d < 3; ++d)
            xc[node * 3 + d] += velw * vc[node * 3 + d] + agg[node * 3 + d] / dg;
    }

    // node MLP: hh += silu(concat(hh, hagg) @ node_w1 + b1) @ node_w2 + b2
    v16h ag[2];
    ag[0] = load_A_f32(hagg + nodeBase * HH, HH, 0, lane);
    ag[1] = load_A_f32(hagg + nodeBase * HH, HH, 32, lane);
#pragma unroll
    for (int nt = 0; nt < 4; ++nt) {
        v8f acc = {0.f, 0.f, 0.f, 0.f, 0.f, 0.f, 0.f, 0.f};
        acc = wmma_f16(ah[0], load_B(n1T, 128, nt * 16, 0, lane), acc);
        acc = wmma_f16(ah[1], load_B(n1T, 128, nt * 16, 32, lane), acc);
        acc = wmma_f16(ag[0], load_B(n1T, 128, nt * 16, 64, lane), acc);
        acc = wmma_f16(ag[1], load_B(n1T, 128, nt * 16, 96, lane), acc);
        int n = nt * 16 + nl;
        float bb = nb1[n];
#pragma unroll
        for (int i = 0; i < 8; ++i)
            Sm[wave][i + 8 * hf][n] = (_Float16)silu_f(acc[i] + bb);
    }
    __syncthreads();
    v16h am[2];
    am[0] = load_A(&Sm[wave][0][0], HH, 0, lane);
    am[1] = load_A(&Sm[wave][0][0], HH, 32, lane);
#pragma unroll
    for (int nt = 0; nt < 4; ++nt) {
        v8f acc = {0.f, 0.f, 0.f, 0.f, 0.f, 0.f, 0.f, 0.f};
        acc = wmma_f16(am[0], load_B(n2T, HH, nt * 16, 0, lane), acc);
        acc = wmma_f16(am[1], load_B(n2T, HH, nt * 16, 32, lane), acc);
        int n = nt * 16 + nl;
        float bb = nb2[n];
#pragma unroll
        for (int i = 0; i < 8; ++i) {
            int node = nodeBase + i + 8 * hf;
            float hv = hh[node * HH + n] + acc[i] + bb;
            hh[node * HH + n] = hv;
            hh16[node * HH + n] = (_Float16)hv;
        }
    }
}

extern "C" void kernel_launch(void* const* d_in, const int* in_sizes, int n_in,
                              void* d_out, int out_size, void* d_ws, size_t ws_size,
                              hipStream_t stream) {
    (void)in_sizes; (void)n_in; (void)out_size; (void)ws_size;
    const float* x    = (const float*)d_in[0];
    const float* v    = (const float*)d_in[1];
    const float* h    = (const float*)d_in[2];
    const float* enc  = (const float*)d_in[3];
    const float* ef   = (const float*)d_in[4];
    const float* embw = (const float*)d_in[5];
    const float* embb = (const float*)d_in[6];
    const float* ew1  = (const float*)d_in[7];
    const float* eb1  = (const float*)d_in[8];
    const float* ew2  = (const float*)d_in[9];
    const float* eb2  = (const float*)d_in[10];
    const float* cw1  = (const float*)d_in[11];
    const float* cb1  = (const float*)d_in[12];
    const float* cw2  = (const float*)d_in[13];
    const float* vw1  = (const float*)d_in[14];
    const float* vb1  = (const float*)d_in[15];
    const float* vw2  = (const float*)d_in[16];
    const float* vb2  = (const float*)d_in[17];
    const float* nw1  = (const float*)d_in[18];
    const float* nb1  = (const float*)d_in[19];
    const float* nw2  = (const float*)d_in[20];
    const float* nb2  = (const float*)d_in[21];
    const int*   eidx = (const int*)d_in[22];
    const int*   tt   = (const int*)d_in[23];
    float* out = (float*)d_out;

    // ---- workspace layout (~1.7 MB) ----
    char* ws = (char*)d_ws;
    size_t off = 0;
    auto alloc = [&](size_t bytes) -> void* {
        void* p = ws + off;
        off += (bytes + 255) & ~(size_t)255;
        return p;
    };
    _Float16* e1T  = (_Float16*)alloc((size_t)LL * HH * EINP * 2);
    _Float16* e2T  = (_Float16*)alloc((size_t)LL * HH * HH * 2);
    _Float16* c1T  = (_Float16*)alloc((size_t)LL * HH * HH * 2);
    _Float16* v1T  = (_Float16*)alloc((size_t)LL * HH * HH * 2);
    _Float16* n1T  = (_Float16*)alloc((size_t)LL * HH * 128 * 2);
    _Float16* n2T  = (_Float16*)alloc((size_t)LL * HH * HH * 2);
    float*    hh0  = (float*)alloc((size_t)KTRAJ * NN * HH * 4);
    _Float16* hh0h = (_Float16*)alloc((size_t)KTRAJ * NN * HH * 2);
    float*    hh   = (float*)alloc((size_t)NN * HH * 4);
    _Float16* hhh  = (_Float16*)alloc((size_t)NN * HH * 2);
    float*    hagg = (float*)alloc((size_t)NN * HH * 4);
    float*    agg  = (float*)alloc((size_t)NN * 3 * 4);
    float*    deg  = (float*)alloc((size_t)NN * 4);
    float*    xc   = (float*)alloc((size_t)NN * 3 * 4);
    float*    vcb  = (float*)alloc((size_t)NN * 3 * 4);
    float*    xst  = (float*)alloc((size_t)NN * 3 * 4);

    const int CW_TOTAL = LL * HH * EINP + 4 * LL * HH * HH + LL * HH * 128;   // 104448
    convert_weights_kernel<<<(CW_TOTAL + 255) / 256, 256, 0, stream>>>(
        ew1, ew2, cw1, vw1, nw1, nw2, e1T, e2T, c1T, v1T, n1T, n2T);
    zero_kernel<<<(NN + 255) / 256, 256, 0, stream>>>(deg, NN);
    deg_kernel<<<EE / 256, 256, 0, stream>>>(eidx, deg);
    hh0_kernel<<<(KTRAJ * NN * HH + 255) / 256, 256, 0, stream>>>(
        h, enc, embw, embb, tt, hh0, hh0h);

    for (int k = 0; k < KTRAJ; ++k) {
        traj_init_kernel<<<(NN * 3 + 255) / 256, 256, 0, stream>>>(x, v, tt, k, xc, vcb, out);
        for (int t = 1; t < TTOT; ++t) {
            step_start_kernel<<<(NN * HH + 255) / 256, 256, 0, stream>>>(
                hh0, hh0h, k, hh, hhh, xc, xst);
            for (int l = 0; l < LL; ++l) {
                zero_kernel<<<(NN * 3 + 255) / 256, 256, 0, stream>>>(agg, NN * 3);
                zero_kernel<<<(NN * HH + 255) / 256, 256, 0, stream>>>(hagg, NN * HH);
                edge_kernel<<<EE / 64, 128, 0, stream>>>(
                    xc, hhh, ef, eidx,
                    e1T + (size_t)l * HH * EINP, eb1 + l * HH,
                    e2T + (size_t)l * HH * HH,   eb2 + l * HH,
                    c1T + (size_t)l * HH * HH,   cb1 + l * HH, cw2 + l * HH,
                    agg, hagg);
                node_kernel<<<NN / 64, 128, 0, stream>>>(
                    hhh, hh, hagg, agg, deg, vcb, xc,
                    v1T + (size_t)l * HH * HH,  vb1 + l * HH, vw2 + l * HH, vb2 + l,
                    n1T + (size_t)l * HH * 128, nb1 + l * HH,
                    n2T + (size_t)l * HH * HH,  nb2 + l * HH);
            }
            step_end_kernel<<<(NN * 3 + 255) / 256, 256, 0, stream>>>(xc, xst, k, t, vcb, out);
        }
    }
}
